// BERTFactorizedAttention_49211735278151
// MI455X (gfx1250) — matrix operand order repaired and checked by reading, not compiled, not run on hardware
//
#include <hip/hip_runtime.h>
#include <hip/hip_bf16.h>

typedef _Float16 f16;
typedef __attribute__((ext_vector_type(16))) _Float16 v16h;
typedef __attribute__((ext_vector_type(8)))  float    v8f;

#define WMMA_F32_F16(a, b, c) \
  __builtin_amdgcn_wmma_f32_16x16x32_f16(false, (a), false, (b), (short)0, (c), false, false)

static constexpr int H  = 1024;  // hidden
static constexpr int NH = 16;    // heads
static constexpr int HD = 64;    // head dim
static constexpr int Bc = 8;     // batch
static constexpr int Sc = 4096;  // seq len

static constexpr int SCH = 8;  // S-chunks for stage-3 split-K
static constexpr int QCH = 4;  // S-chunks for stage-2 stats

// ---------------------------------------------------------------------------
// Async global->LDS via inline asm (gfx1250 GLOBAL_LOAD_ASYNC_TO_LDS_B128,
// tracked by ASYNCcnt).  GV addressing: 64-bit global addr in a VGPR pair,
// wave-relative LDS byte offset in vdst (low 32 bits of flat AS3 pointer).
// ---------------------------------------------------------------------------
__device__ __forceinline__ void async_b128(void* lds, const void* g) {
  const unsigned int l = (unsigned int)(size_t)lds;
  asm volatile("global_load_async_to_lds_b128 %0, %1, off"
               :
               : "v"(l), "v"(g)
               : "memory");
}

__device__ __forceinline__ void async_wait_all() {
  asm volatile("s_wait_asynccnt 0" ::: "memory");
}

// ---------------------------------------------------------------------------
// WMMA fragment loaders from LDS (CDNA5 16-bit layouts, CDNA5 ISA)
// ---------------------------------------------------------------------------
__device__ __forceinline__ v16h load_frag_a(const f16* lds, int ldr) {
  const int lane = threadIdx.x & 31;
  const f16* row = lds + (lane & 15) * ldr + (lane >> 4) * 8;
  v16h a;
#pragma unroll
  for (int j = 0; j < 8; ++j) {
    const int k = (j < 4) ? (2 * j) : (16 + 2 * (j - 4));
    a[2 * j]     = row[k];
    a[2 * j + 1] = row[k + 1];
  }
  return a;
}

__device__ __forceinline__ v16h load_frag_a_f32(const float* lds, int ldr) {
  const int lane   = threadIdx.x & 31;
  const float* row = lds + (lane & 15) * ldr + (lane >> 4) * 8;
  v16h a;
#pragma unroll
  for (int j = 0; j < 8; ++j) {
    const int k = (j < 4) ? (2 * j) : (16 + 2 * (j - 4));
    a[2 * j]     = (f16)row[k];
    a[2 * j + 1] = (f16)row[k + 1];
  }
  return a;
}

__device__ __forceinline__ v16h load_frag_b_t(const f16* lds, int ldr) {
  const int lane = threadIdx.x & 31;
  const f16* col = lds + (lane & 15) * ldr + (lane >> 4) * 8;
  v16h b;
#pragma unroll
  for (int j = 0; j < 8; ++j) {
    const int k = (j < 4) ? (2 * j) : (16 + 2 * (j - 4));
    b[2 * j]     = col[k];
    b[2 * j + 1] = col[k + 1];
  }
  return b;
}

// ---------------------------------------------------------------------------
// Kernel 1: fused QKV projection.  out = (f16)(hidden @ W + bias)
// grid = (H/64, M/128, 3), block = 256 (8 waves).
// A staged fp32 via async DMA, double-buffered, 1 barrier per K-step.
// ---------------------------------------------------------------------------
#define BM 128
#define BN 64
#define BK 32
#define PAD 8
#define APADF 4

__global__ __launch_bounds__(256) void qkv_gemm_kernel(
    const float* __restrict__ hidden,
    const float* __restrict__ Wq, const float* __restrict__ bq,
    const float* __restrict__ Wk, const float* __restrict__ bk,
    const float* __restrict__ Wv, const float* __restrict__ bv,
    f16* __restrict__ qout, f16* __restrict__ kout, f16* __restrict__ vout) {
  const float* W;
  const float* bias;
  f16* out;
  if (blockIdx.z == 0)      { W = Wq; bias = bq; out = qout; }
  else if (blockIdx.z == 1) { W = Wk; bias = bk; out = kout; }
  else                      { W = Wv; bias = bv; out = vout; }

  __shared__ float Af[2][BM][BK + APADF];  // fp32 A tiles (async-staged)
  __shared__ f16   Bt[2][BN][BK + PAD];    // W^T tiles, f16
  __shared__ f16   Cs[BM][BN + 8];         // output staging

  const int t      = threadIdx.x;
  const int wave   = t >> 5;
  const int lane   = t & 31;
  const int c      = lane & 15;
  const int hlf    = lane >> 4;
  const int mBlock = blockIdx.y * BM;
  const int nBlock = blockIdx.x * BN;

  const int ach = t & 7;          // 8 x 16B chunks per 32-float A row
  const int ar0 = t >> 3;         // 32 rows per pass
  const int wn  = t & 63;
  const int wk0 = (t >> 6) * 2;   // adjacent k pairs -> ds_store_b32 merge

  v8f acc[4] = {v8f{}, v8f{}, v8f{}, v8f{}};

  auto stageA = [&](int buf, int k0) {
#pragma unroll
    for (int p = 0; p < 4; ++p) {
      const int r = ar0 + p * 32;
      async_b128(&Af[buf][r][ach * 4],
                 hidden + (size_t)(mBlock + r) * H + k0 + ach * 4);
    }
  };
  auto loadW = [&](int k0, float* wreg) {
#pragma unroll
    for (int p = 0; p < 4; ++p) {
      wreg[2 * p]     = W[(size_t)(k0 + wk0 + p * 8) * H + nBlock + wn];
      wreg[2 * p + 1] = W[(size_t)(k0 + wk0 + p * 8 + 1) * H + nBlock + wn];
    }
  };
  auto storeW = [&](int buf, const float* wreg) {
#pragma unroll
    for (int p = 0; p < 4; ++p) {
      Bt[buf][wn][wk0 + p * 8]     = (f16)wreg[2 * p];
      Bt[buf][wn][wk0 + p * 8 + 1] = (f16)wreg[2 * p + 1];
    }
  };

  // prologue: stage tile 0
  stageA(0, 0);
  {
    float wreg[8];
    loadW(0, wreg);
    storeW(0, wreg);
  }
  async_wait_all();
  __syncthreads();

  const int NK = H / BK;  // 32
  for (int kt = 0; kt < NK; ++kt) {
    const int cur = kt & 1;
    const int nxt = cur ^ 1;
    float wreg[8];
    const bool more = (kt + 1 < NK);
    if (more) {
      stageA(nxt, (kt + 1) * BK);   // async DMA for next tile
      loadW((kt + 1) * BK, wreg);   // W for next tile into regs
    }

    const v16h af = load_frag_a_f32(&Af[cur][wave * 16][0], BK + APADF);
#pragma unroll
    for (int nt = 0; nt < 4; ++nt) {
      const v16h bf = load_frag_b_t(&Bt[cur][nt * 16][0], BK + PAD);
      acc[nt]       = WMMA_F32_F16(af, bf, acc[nt]);
    }

    if (more) storeW(nxt, wreg);
    async_wait_all();
    __syncthreads();
  }

  // epilogue: bias add, stage to LDS, coalesced b128 stores
#pragma unroll
  for (int nt = 0; nt < 4; ++nt) {
    const int n    = nt * 16 + c;
    const float bz = bias[nBlock + n];
#pragma unroll
    for (int i = 0; i < 8; ++i)
      Cs[wave * 16 + hlf * 8 + i][n] = (f16)(acc[nt][i] + bz);
  }
  __syncthreads();
#pragma unroll
  for (int p = 0; p < 4; ++p) {
    const int r = ar0 + p * 32;
    *(float4*)(out + (size_t)(mBlock + r) * H + nBlock + ach * 8) =
        *(const float4*)(&Cs[r][ach * 8]);
  }
}

// ---------------------------------------------------------------------------
// Kernel 2a: partial per-column online-softmax stats of q (+mask).
// grid = (B*H/64, QCH); each block: 64 columns, one S-chunk.
// ---------------------------------------------------------------------------
__global__ __launch_bounds__(256) void col_stats_partial_kernel(
    const f16* __restrict__ q, const float* __restrict__ mask,
    float* __restrict__ pmax, float* __restrict__ psum) {
  const int b     = blockIdx.x / (H / 64);
  const int cBase = (blockIdx.x % (H / 64)) * 64;
  const int chunk = blockIdx.y;
  const int t     = threadIdx.x;
  const int col   = cBase + (t & 63);
  const int rg    = t >> 6;
  const int r0    = chunk * (Sc / QCH);

  float m = -3.0e38f, s = 0.f;
  for (int r = r0 + rg; r < r0 + Sc / QCH; r += 4) {
    const float x =
        (float)q[((size_t)b * Sc + r) * H + col] + mask[(size_t)b * Sc + r];
    const float mn = fmaxf(m, x);
    s = s * __expf(m - mn) + __expf(x - mn);
    m = mn;
  }
  __shared__ float sm[4][64];
  __shared__ float ss[4][64];
  sm[rg][t & 63] = m;
  ss[rg][t & 63] = s;
  __syncthreads();
  if (rg == 0) {
#pragma unroll
    for (int i = 1; i < 4; ++i) {
      const float mo = sm[i][t & 63], so = ss[i][t & 63];
      const float mn = fmaxf(m, mo);
      s = s * __expf(m - mn) + so * __expf(mo - mn);
      m = mn;
    }
    pmax[(size_t)chunk * Bc * H + (size_t)b * H + col] = m;
    psum[(size_t)chunk * Bc * H + (size_t)b * H + col] = s;
  }
}

// Kernel 2b: merge the QCH partial (max,sum) pairs.  grid = B*H/256.
__global__ __launch_bounds__(256) void col_stats_merge_kernel(
    const float* __restrict__ pmax, const float* __restrict__ psum,
    float* __restrict__ colmax, float* __restrict__ colsum) {
  const int i = blockIdx.x * 256 + threadIdx.x;  // < B*H
  float m = pmax[i], s = psum[i];
#pragma unroll
  for (int cck = 1; cck < QCH; ++cck) {
    const float mo = pmax[(size_t)cck * Bc * H + i];
    const float so = psum[(size_t)cck * Bc * H + i];
    const float mn = fmaxf(m, mo);
    s = s * __expf(m - mn) + so * __expf(mo - mn);
    m = mn;
  }
  colmax[i] = m;
  colsum[i] = s;
}

// ---------------------------------------------------------------------------
// Kernel 3: partial s_ctx (64x64 fp32) = P^T @ V over one S-chunk.
// grid = (SCH, B*NH), block = 256.  Double-buffered, 1 barrier per K-step.
// ---------------------------------------------------------------------------
__global__ __launch_bounds__(256) void sctx_gemm_kernel(
    const f16* __restrict__ q, const f16* __restrict__ v,
    const float* __restrict__ mask, const float* __restrict__ colmax,
    const float* __restrict__ colsum, float* __restrict__ sctxP) {
  const int chunk = blockIdx.x;        // S-chunk
  const int bh    = blockIdx.y;        // b*NH + h
  const int b     = bh >> 4;
  const int hd    = (bh & (NH - 1)) * HD;
  const int sOff  = chunk * (Sc / SCH);

  __shared__ f16 Pt[2][64][BK + PAD];  // probs^T [d][s_local]
  __shared__ f16 Vt[2][64][BK + PAD];  // v^T     [e][s_local]

  const int t    = threadIdx.x;
  const int wave = t >> 5;
  const int lane = t & 31;
  const int c    = lane & 15;
  const int hlf  = lane >> 4;
  const int mSub = wave & 3;
  const int nhh  = wave >> 2;

  const int d   = t & 63;
  const int sl0 = t >> 6;

  const float cmax = colmax[(size_t)b * H + hd + d];
  const float rinv = 1.f / colsum[(size_t)b * H + hd + d];

  auto stage = [&](int buf, int s0) {
#pragma unroll
    for (int p = 0; p < 8; ++p) {
      const int sl     = sl0 + p * 4;
      const size_t row = (size_t)b * Sc + sOff + s0 + sl;
      const float qv   = (float)q[row * H + hd + d];
      Pt[buf][d][sl]   = (f16)(__expf(qv + mask[row] - cmax) * rinv);
      Vt[buf][d][sl]   = v[row * H + hd + d];
    }
  };

  v8f acc[2] = {v8f{}, v8f{}};

  stage(0, 0);
  __syncthreads();

  const int NKS = (Sc / SCH) / BK;  // 16
  for (int kt = 0; kt < NKS; ++kt) {
    const int cur = kt & 1;
    const int nxt = cur ^ 1;
    if (kt + 1 < NKS) stage(nxt, (kt + 1) * BK);

    const v16h af = load_frag_a(&Pt[cur][mSub * 16][0], BK + PAD);
#pragma unroll
    for (int j = 0; j < 2; ++j) {
      const v16h bf = load_frag_b_t(&Vt[cur][(nhh * 2 + j) * 16][0], BK + PAD);
      acc[j]        = WMMA_F32_F16(af, bf, acc[j]);
    }
    __syncthreads();
  }

  // partial fp32 tile, natural [d][e] order
  float* o = sctxP + ((size_t)chunk * Bc * NH + bh) * (size_t)(HD * HD);
#pragma unroll
  for (int j = 0; j < 2; ++j) {
    const int n = (nhh * 2 + j) * 16 + c;  // e
#pragma unroll
    for (int i = 0; i < 8; ++i) {
      const int mr = mSub * 16 + hlf * 8 + i;  // d
      o[mr * HD + n] = acc[j][i];
    }
  }
}

// Kernel 3b: deterministic reduce of SCH partials, convert to f16 and
// TRANSPOSE ([e][d]) for kernel 4's async B-operand copy.  grid = BH*4096/256.
__global__ __launch_bounds__(256) void sctx_reduce_kernel(
    const float* __restrict__ sctxP, f16* __restrict__ sctxT) {
  const int idx = blockIdx.x * 256 + threadIdx.x;  // < B*NH*64*64
  const int bh  = idx >> 12;
  const int r   = idx & 4095;
  const int d   = r >> 6;
  const int e   = r & 63;
  float s = 0.f;
#pragma unroll
  for (int cck = 0; cck < SCH; ++cck)
    s += sctxP[((size_t)cck * Bc * NH + bh) * 4096 + r];
  sctxT[(size_t)bh * 4096 + e * HD + d] = (f16)s;
}

// ---------------------------------------------------------------------------
// Kernel 4: ctx = rowsoftmax(k_head) (128x64) @ s_ctx (64x64) -> fp32 out.
// s_ctx^T async-copied into LDS (overlaps the softmax VALU phase);
// output staged through LDS for coalesced b128 stores.
// ---------------------------------------------------------------------------
__global__ __launch_bounds__(256) void ctx_kernel(
    const f16* __restrict__ k, const f16* __restrict__ sctxT,
    float* __restrict__ out) {
  const int b     = blockIdx.z;
  const int h     = blockIdx.y;
  const int sBase = blockIdx.x * 128;
  const int hd    = h * HD;

  __shared__ f16   Sct[64][64 + 8];   // s_ctx^T [e][d]  (B operand layout)
  __shared__ f16   cp[128][64 + 8];   // c_probs [row][d]
  __shared__ float Co[128][64 + 4];   // output staging

  const int t = threadIdx.x;

  // async DMA of s_ctx^T (64x64 f16, per-lane 16B chunks, pad-aware)
  const f16* scT = sctxT + ((size_t)b * NH + h) * (size_t)(HD * HD);
#pragma unroll
  for (int p = 0; p < 2; ++p) {
    const int i  = t + p * 256;  // 0..511
    const int e  = i >> 3;
    const int ch = i & 7;
    async_b128(&Sct[e][ch * 8], scT + (size_t)e * HD + ch * 8);
  }

  // row softmax of k_head for 128 rows (threads 0..127, one row each)
  if (t < 128) {
    const f16* krow = k + ((size_t)b * Sc + sBase + t) * H + hd;
    float xv[64];
    float m = -3.0e38f;
#pragma unroll
    for (int j = 0; j < 64; ++j) {
      xv[j] = (float)krow[j];
      m     = fmaxf(m, xv[j]);
    }
    float s = 0.f;
#pragma unroll
    for (int j = 0; j < 64; ++j) {
      xv[j] = __expf(xv[j] - m);
      s += xv[j];
    }
    const float rinv = 1.f / s;
#pragma unroll
    for (int j = 0; j < 64; ++j) cp[t][j] = (f16)(xv[j] * rinv);
  }
  async_wait_all();
  __syncthreads();

  const int wave = t >> 5;
  const int lane = t & 31;
  const int c    = lane & 15;
  const int hlf  = lane >> 4;

  v8f acc[4] = {v8f{}, v8f{}, v8f{}, v8f{}};
#pragma unroll
  for (int ks = 0; ks < 2; ++ks) {
    const v16h af = load_frag_a(&cp[wave * 16][ks * 32], 64 + 8);
#pragma unroll
    for (int nt = 0; nt < 4; ++nt) {
      const v16h bf = load_frag_b_t(&Sct[nt * 16][ks * 32], 64 + 8);
      acc[nt]       = WMMA_F32_F16(af, bf, acc[nt]);
    }
  }

  // stage fp32 result, then coalesced stores
#pragma unroll
  for (int nt = 0; nt < 4; ++nt) {
    const int n = nt * 16 + c;
#pragma unroll
    for (int i = 0; i < 8; ++i) Co[wave * 16 + hlf * 8 + i][n] = acc[nt][i];
  }
  __syncthreads();
#pragma unroll
  for (int p = 0; p < 8; ++p) {
    const int r  = (t >> 4) + p * 16;
    const int ch = t & 15;  // 16 x 16B chunks per 64-float row
    *(float4*)(out + ((size_t)b * Sc + sBase + r) * H + hd + ch * 4) =
        *(const float4*)(&Co[r][ch * 4]);
  }
}

// ---------------------------------------------------------------------------
// Launch
// ---------------------------------------------------------------------------
extern "C" void kernel_launch(void* const* d_in, const int* in_sizes, int n_in,
                              void* d_out, int out_size, void* d_ws,
                              size_t ws_size, hipStream_t stream) {
  const float* hidden = (const float*)d_in[0];
  const float* mask   = (const float*)d_in[1];
  const float* Wq     = (const float*)d_in[2];
  const float* bq     = (const float*)d_in[3];
  const float* Wk     = (const float*)d_in[4];
  const float* bk     = (const float*)d_in[5];
  const float* Wv     = (const float*)d_in[6];
  const float* bv     = (const float*)d_in[7];
  float* out          = (float*)d_out;

  const int M = Bc * Sc;  // 32768 rows

  // workspace layout
  char* ws              = (char*)d_ws;
  const size_t qkvBytes = (size_t)M * H * sizeof(f16);  // 64 MB each
  f16* qbuf     = (f16*)(ws);
  f16* kbuf     = (f16*)(ws + qkvBytes);
  f16* vbuf     = (f16*)(ws + 2 * qkvBytes);
  char* p       = ws + 3 * qkvBytes;
  float* colmax = (float*)p;                 p += (size_t)Bc * H * 4;
  float* colsum = (float*)p;                 p += (size_t)Bc * H * 4;
  float* pmax   = (float*)p;                 p += (size_t)QCH * Bc * H * 4;
  float* psum   = (float*)p;                 p += (size_t)QCH * Bc * H * 4;
  float* sctxP  = (float*)p;                 p += (size_t)SCH * Bc * NH * HD * HD * 4;
  f16* sctxT    = (f16*)p;

  // 1) fused QKV projection (f16 WMMA, fp32 accumulate, async-LDS A staging)
  qkv_gemm_kernel<<<dim3(H / BN, M / BM, 3), 256, 0, stream>>>(
      hidden, Wq, bq, Wk, bk, Wv, bv, qbuf, kbuf, vbuf);

  // 2) column softmax stats for q (+mask), split over S then merged
  col_stats_partial_kernel<<<dim3(Bc * (H / 64), QCH), 256, 0, stream>>>(
      qbuf, mask, pmax, psum);
  col_stats_merge_kernel<<<(Bc * H) / 256, 256, 0, stream>>>(pmax, psum,
                                                             colmax, colsum);

  // 3) s_ctx partials (split-K over S), then reduce + transpose to f16
  sctx_gemm_kernel<<<dim3(SCH, Bc * NH), 256, 0, stream>>>(
      qbuf, vbuf, mask, colmax, colsum, sctxP);
  sctx_reduce_kernel<<<(Bc * NH * HD * HD) / 256, 256, 0, stream>>>(sctxP,
                                                                    sctxT);

  // 4) ctx = rowsoftmax(K) @ s_ctx, fp32 output [B,S,H]
  ctx_kernel<<<dim3(Sc / 128, NH, Bc), 256, 0, stream>>>(kbuf, sctxT, out);
}
